// GRU_77867757076552
// MI455X (gfx1250) — matrix operand order repaired and checked by reading, compiled, not run on hardware
//
#include <hip/hip_runtime.h>
#include <hip/hip_bf16.h>

// ---------------------------------------------------------------------------
// Types for CDNA5 WMMA (wave32): bf16 A/B fragments, f32 accumulator
// ---------------------------------------------------------------------------
typedef __attribute__((ext_vector_type(16))) __bf16 v16bf;
typedef __attribute__((ext_vector_type(8)))  __bf16 v8bf;
typedef __attribute__((ext_vector_type(8)))  float  v8f;
typedef __attribute__((ext_vector_type(4)))  float  v4f;
typedef __attribute__((ext_vector_type(4)))  unsigned int u32x4;
typedef __attribute__((ext_vector_type(8)))  int          i32x8;
typedef __attribute__((ext_vector_type(4)))  int          i32x4;

#define LDS_STRIDE 40            // 32 bf16 data + 8 bf16 pad per row (80B) -> no bank conflicts
#define TILE_ELEMS (16 * LDS_STRIDE)

__device__ __forceinline__ v16bf bf_cat(v8bf lo, v8bf hi) {
    return __builtin_shufflevector(lo, hi, 0,1,2,3,4,5,6,7,8,9,10,11,12,13,14,15);
}

__device__ __forceinline__ v8f wmma_bf16(v16bf a, v16bf b, v8f c) {
    return __builtin_amdgcn_wmma_f32_16x16x32_bf16(
        false, a, false, b, (short)0, c, false, false);
}

// A fragment from an LDS 16x32 tile (contiguous k: lane half h holds k 16h..16h+15).
__device__ __forceinline__ v16bf lds_a_frag(const __bf16* t, int lane) {
    const int row = lane & 15;
    const int kb  = (lane >> 4) << 4;          // 0 or 16
    const __bf16* p = t + row * LDS_STRIDE + kb;
    v8bf lo = *(const v8bf*)(p);
    v8bf hi = *(const v8bf*)(p + 8);
    return bf_cat(lo, hi);
}

// B fragment (B = W^T, 32x16): lane -> col n0+(lane&15), 16 consecutive K.
__device__ __forceinline__ v16bf b_frag_bf16(const __bf16* __restrict__ W,
                                             int n0, int k0, int lane) {
    const int n = n0 + (lane & 15);
    const int k = k0 + ((lane >> 4) << 4);
    const __bf16* p = W + (size_t)n * 1024 + k;
    v8bf lo = *(const v8bf*)(p);
    v8bf hi = *(const v8bf*)(p + 8);
    return bf_cat(lo, hi);
}

// Same, converting fp32 weights (W_out) on the fly.
__device__ __forceinline__ v16bf b_frag_f32(const float* __restrict__ W,
                                            int n0, int k0, int lane) {
    const int n = n0 + (lane & 15);
    const int k = k0 + ((lane >> 4) << 4);
    const float* p = W + (size_t)n * 1024 + k;
    v16bf r;
#pragma unroll
    for (int i = 0; i < 16; i += 4) {
        v4f f = *(const v4f*)(p + i);
        r[i + 0] = (__bf16)f[0];
        r[i + 1] = (__bf16)f[1];
        r[i + 2] = (__bf16)f[2];
        r[i + 3] = (__bf16)f[3];
    }
    return r;
}

// ---------------------------------------------------------------------------
// Tensor Data Mover: load a 128(rows) x 32(cols) bf16 tile from a row-major
// (4096 x 1024) tensor into LDS, inserting 16B padding after every 64B row
// (-> LDS row stride 80B = LDS_STRIDE bf16).  D# per ISA ch.8.
// 6-arg clang-23 builtin form.
// ---------------------------------------------------------------------------
__device__ __forceinline__ void tdm_load_tile(const __bf16* gsrc, unsigned lds_off) {
    const unsigned long long ga = (unsigned long long)(size_t)gsrc;
    u32x4 g0;
    g0[0] = 1u;                                        // count=1 (valid), user mode
    g0[1] = lds_off;                                   // lds_addr (bytes)
    g0[2] = (unsigned)(ga & 0xFFFFFFFFu);              // global_addr[31:0]
    g0[3] = (unsigned)((ga >> 32) & 0x1FFFFFFu)        // global_addr[56:32]
          | (2u << 30);                                // type=2 ("image")
    i32x8 g1;
    g1[0] = (1 << 16)                                  // data_size = 2 bytes
          | (1 << 20)                                  // pad_enable
          | (3 << 22)                                  // pad_interval: 16 DWORDs (64B)
          | (3 << 25);                                 // pad_amount: 4 DWORDs (16B)
    g1[1] = (int)(1024u << 16);                        // tensor_dim0 = 1024
    g1[2] = (int)(4096u << 16);                        // tensor_dim1 = 4096
    g1[3] = (int)(32u << 16);                          // tile_dim0 = 32
    g1[4] = 128;                                       // tile_dim1 = 128 (tile_dim2 = 0)
    g1[5] = 1024;                                      // tensor_dim0_stride = 1024
    g1[6] = 0;
    g1[7] = 0;
    const i32x4 z4 = {0, 0, 0, 0};                     // groups 2/3 unused (2D)
    const i32x8 z8 = {0, 0, 0, 0, 0, 0, 0, 0};
    __builtin_amdgcn_tensor_load_to_lds(g0, g1, z4, z4, z8, 0);
}

// ---------------------------------------------------------------------------
// Pre-pass kernels
// ---------------------------------------------------------------------------
__global__ void __launch_bounds__(256)
cvt_f32_bf16(const float* __restrict__ s, __bf16* __restrict__ d, int n) {
    int i = (blockIdx.x * 256 + threadIdx.x) * 4;
    if (i + 3 < n) {
        v4f f = *(const v4f*)(s + i);
        d[i + 0] = (__bf16)f[0];
        d[i + 1] = (__bf16)f[1];
        d[i + 2] = (__bf16)f[2];
        d[i + 3] = (__bf16)f[3];
    }
}

__global__ void __launch_bounds__(256)
embed_bf16(const int* __restrict__ idx, const float* __restrict__ emb,
           __bf16* __restrict__ x) {
    const int sb = blockIdx.x;
    const int tok = idx[sb];
    const float* src = emb + (size_t)tok * 1024;
    __bf16* dst = x + (size_t)sb * 1024;
    for (int i = threadIdx.x; i < 1024; i += 256) dst[i] = (__bf16)src[i];
}

// ---------------------------------------------------------------------------
// GRU phase 1 (double-buffered LDS; split accumulators so the 5 WMMAs per
// k-step are mutually independent -> no WMMA->WMMA hazard NOPs)
// ---------------------------------------------------------------------------
__global__ void __launch_bounds__(128)
gru_gates(const __bf16* __restrict__ inp, const float* __restrict__ h,
          const __bf16* __restrict__ Wrx, const __bf16* __restrict__ Wrh,
          const __bf16* __restrict__ Wzx, const __bf16* __restrict__ Wzh,
          const __bf16* __restrict__ Whx,
          const float* __restrict__ br, const float* __restrict__ bz,
          const float* __restrict__ bh,
          float* __restrict__ r_o, float* __restrict__ z_o,
          float* __restrict__ xh_o) {
    __shared__ __bf16 sI[2][TILE_ELEMS];
    __shared__ __bf16 sH[2][TILE_ELEMS];
    const int tid  = threadIdx.x;
    const int lane = tid & 31;
    const int wave = tid >> 5;
    const int n0 = (blockIdx.x * 4 + wave) * 16;
    const int m0 = blockIdx.y * 16;
    const int srow = tid >> 3;            // 0..15
    const int scol = (tid & 7) * 4;       // 0,4,...,28
    const __bf16* iprow = inp + (size_t)(m0 + srow) * 1024 + scol;
    const float*  hprow = h   + (size_t)(m0 + srow) * 1024 + scol;
    const int sdst = srow * LDS_STRIDE + scol;

    auto stage = [&](int buf, int kt) {
        const __bf16* ip = iprow + kt * 32;
        const float*  hp = hprow + kt * 32;
#pragma unroll
        for (int i = 0; i < 4; ++i) {
            sI[buf][sdst + i] = ip[i];
            sH[buf][sdst + i] = (__bf16)hp[i];
        }
    };

    // independent partial sums: *0 accumulates inp@W*x^T, *1 accumulates h@W*h^T
    v8f cr0 = {}, cr1 = {}, cz0 = {}, cz1 = {}, ch0 = {};
    stage(0, 0);
    for (int kt = 0; kt < 32; ++kt) {
        __syncthreads();                      // prev stage stores + prev reads done
        const int cur = kt & 1;
        if (kt + 1 < 32) stage(cur ^ 1, kt + 1);
        const int k0 = kt * 32;
        v16bf aI = lds_a_frag(sI[cur], lane);
        v16bf aH = lds_a_frag(sH[cur], lane);
        cr0 = wmma_bf16(aI, b_frag_bf16(Wrx, n0, k0, lane), cr0);
        cz0 = wmma_bf16(aI, b_frag_bf16(Wzx, n0, k0, lane), cz0);
        ch0 = wmma_bf16(aI, b_frag_bf16(Whx, n0, k0, lane), ch0);
        cr1 = wmma_bf16(aH, b_frag_bf16(Wrh, n0, k0, lane), cr1);
        cz1 = wmma_bf16(aH, b_frag_bf16(Wzh, n0, k0, lane), cz1);
    }

    const int col  = n0 + (lane & 15);
    const int rowB = (lane >> 4) * 8;
    const float brv = br[col], bzv = bz[col], bhv = bh[col];
#pragma unroll
    for (int j = 0; j < 8; ++j) {
        const size_t idx = (size_t)(m0 + rowB + j) * 1024 + col;
        r_o[idx]  = 1.0f / (1.0f + __expf(-(cr0[j] + cr1[j] + brv)));
        z_o[idx]  = 1.0f / (1.0f + __expf(-(cz0[j] + cz1[j] + bzv)));
        xh_o[idx] = ch0[j] + bhv;
    }
}

// ---------------------------------------------------------------------------
// GRU phase 2 (double-buffered LDS)
// ---------------------------------------------------------------------------
__global__ void __launch_bounds__(128)
gru_combine(const float* __restrict__ h, const float* __restrict__ r,
            const float* __restrict__ z, const float* __restrict__ xh,
            const __bf16* __restrict__ Whh,
            float* __restrict__ h_out, __bf16* __restrict__ next_bf) {
    __shared__ __bf16 sHR[2][TILE_ELEMS];
    const int tid  = threadIdx.x;
    const int lane = tid & 31;
    const int wave = tid >> 5;
    const int n0 = (blockIdx.x * 4 + wave) * 16;
    const int m0 = blockIdx.y * 16;
    const int srow = tid >> 3;
    const int scol = (tid & 7) * 4;
    const float* hprow = h + (size_t)(m0 + srow) * 1024 + scol;
    const float* rprow = r + (size_t)(m0 + srow) * 1024 + scol;
    const int sdst = srow * LDS_STRIDE + scol;

    auto stage = [&](int buf, int kt) {
        const float* hp = hprow + kt * 32;
        const float* rp = rprow + kt * 32;
#pragma unroll
        for (int i = 0; i < 4; ++i)
            sHR[buf][sdst + i] = (__bf16)(hp[i] * rp[i]);
    };

    // two interleaved partial sums to break the per-iteration D->C chain
    v8f c0 = {}, c1 = {};
    stage(0, 0);
    for (int kt = 0; kt < 32; kt += 2) {
        {
            __syncthreads();
            const int cur = kt & 1;
            stage(cur ^ 1, kt + 1);
            c0 = wmma_bf16(lds_a_frag(sHR[cur], lane),
                           b_frag_bf16(Whh, n0, kt * 32, lane), c0);
        }
        {
            __syncthreads();
            const int cur = (kt + 1) & 1;
            if (kt + 2 < 32) stage(cur ^ 1, kt + 2);
            c1 = wmma_bf16(lds_a_frag(sHR[cur], lane),
                           b_frag_bf16(Whh, n0, (kt + 1) * 32, lane), c1);
        }
    }

    const int col  = n0 + (lane & 15);
    const int rowB = (lane >> 4) * 8;
#pragma unroll
    for (int j = 0; j < 8; ++j) {
        const size_t idx = (size_t)(m0 + rowB + j) * 1024 + col;
        const float ht = tanhf(xh[idx] + c0[j] + c1[j]);
        const float zv = z[idx];
        const float hn = (1.0f - zv) * h[idx] + zv * ht;
        h_out[idx]   = hn;
        next_bf[idx] = (__bf16)hn;
    }
}

// ---------------------------------------------------------------------------
// Output projection with TDM-staged A tiles (double-buffered) and NT stores.
// ---------------------------------------------------------------------------
__global__ void __launch_bounds__(128)
gru_logits(const __bf16* __restrict__ A, const float* __restrict__ Wo,
           const float* __restrict__ bo, float* __restrict__ out) {
    __shared__ __bf16 sA[2][8 * TILE_ELEMS];           // 2 x 10 KB
    const int tid  = threadIdx.x;
    const int lane = tid & 31;
    const int wave = tid >> 5;
    const int n0    = (blockIdx.x * 4 + wave) * 16;
    const int mbase = blockIdx.y * 128;
    const bool issuer = (wave == 0);
    const __bf16* slab = A + (size_t)mbase * 1024;     // 128 rows of A

    if (issuer) tdm_load_tile(slab, (unsigned)(size_t)&sA[0][0]);
    v8f acc[8] = {};

    for (int kt = 0; kt < 32; ++kt) {
        __syncthreads();                               // reads of next buf finished
        if (issuer) {
            if (kt + 1 < 32) {
                tdm_load_tile(slab + (kt + 1) * 32,
                              (unsigned)(size_t)&sA[(kt + 1) & 1][0]);
                __builtin_amdgcn_s_wait_tensorcnt(1);  // tile kt landed
            } else {
                __builtin_amdgcn_s_wait_tensorcnt(0);
            }
        }
        __syncthreads();                               // release waves 1..3
        v16bf b = b_frag_f32(Wo, n0, kt * 32, lane);
        const __bf16* bufp = sA[kt & 1];
#pragma unroll
        for (int mt = 0; mt < 8; ++mt)
            acc[mt] = wmma_bf16(lds_a_frag(bufp + mt * TILE_ELEMS, lane), b, acc[mt]);
    }

    const int col  = n0 + (lane & 15);
    const int rowB = (lane >> 4) * 8;
    const float bov = bo[col];
#pragma unroll
    for (int mt = 0; mt < 8; ++mt) {
#pragma unroll
        for (int j = 0; j < 8; ++j) {
            const int row = mbase + mt * 16 + rowB + j;
            // streaming 512MB: NT so we don't evict L2-resident W_out
            __builtin_nontemporal_store(acc[mt][j] + bov,
                                        out + (size_t)row * 32000 + col);
        }
    }
}

// ---------------------------------------------------------------------------
// Host orchestration
// ---------------------------------------------------------------------------
extern "C" void kernel_launch(void* const* d_in, const int* in_sizes, int n_in,
                              void* d_out, int out_size, void* d_ws, size_t ws_size,
                              hipStream_t stream) {
    (void)in_sizes; (void)n_in; (void)out_size; (void)ws_size;
    constexpr int S = 64, B = 64, H = 1024, V = 32000, L = 2;
    constexpr size_t HH = (size_t)H * H;
    constexpr size_t BH = (size_t)B * H;

    const int*   inputs = (const int*)  d_in[0];
    const float* hidden = (const float*)d_in[1];
    const float* emb    = (const float*)d_in[2];
    const float* Wrx_f  = (const float*)d_in[3];
    const float* br     = (const float*)d_in[4];
    const float* Wrh_f  = (const float*)d_in[5];
    const float* Wzx_f  = (const float*)d_in[6];
    const float* bz     = (const float*)d_in[7];
    const float* Wzh_f  = (const float*)d_in[8];
    const float* Whx_f  = (const float*)d_in[9];
    const float* bh     = (const float*)d_in[10];
    const float* Whh_f  = (const float*)d_in[11];
    const float* Wout   = (const float*)d_in[12];
    const float* bout   = (const float*)d_in[13];
    float* out = (float*)d_out;

    char* base = (char*)d_ws;
    size_t off = 0;
    auto take = [&](size_t bytes) -> char* {
        char* p = base + off;
        off += (bytes + 255) & ~(size_t)255;
        return p;
    };
    __bf16* Wrx = (__bf16*)take(L * HH * 2);
    __bf16* Wrh = (__bf16*)take(L * HH * 2);
    __bf16* Wzx = (__bf16*)take(L * HH * 2);
    __bf16* Wzh = (__bf16*)take(L * HH * 2);
    __bf16* Whx = (__bf16*)take(L * HH * 2);
    __bf16* Whh = (__bf16*)take(L * HH * 2);
    __bf16* xbf   = (__bf16*)take((size_t)S * BH * 2);
    __bf16* topbf = (__bf16*)take((size_t)S * BH * 2);
    __bf16* inp1  = (__bf16*)take(BH * 2);
    float*  hbuf  = (float*)take(2 * L * BH * 4);
    float*  rbuf  = (float*)take(BH * 4);
    float*  zbuf  = (float*)take(BH * 4);
    float*  xhbuf = (float*)take(BH * 4);

    const int nW = (int)(L * HH);
    cvt_f32_bf16<<<nW / 1024, 256, 0, stream>>>(Wrx_f, Wrx, nW);
    cvt_f32_bf16<<<nW / 1024, 256, 0, stream>>>(Wrh_f, Wrh, nW);
    cvt_f32_bf16<<<nW / 1024, 256, 0, stream>>>(Wzx_f, Wzx, nW);
    cvt_f32_bf16<<<nW / 1024, 256, 0, stream>>>(Wzh_f, Wzh, nW);
    cvt_f32_bf16<<<nW / 1024, 256, 0, stream>>>(Whx_f, Whx, nW);
    cvt_f32_bf16<<<nW / 1024, 256, 0, stream>>>(Whh_f, Whh, nW);

    embed_bf16<<<S * B, 256, 0, stream>>>(inputs, emb, xbf);

    (void)hipMemcpyAsync(hbuf, hidden, L * BH * sizeof(float),
                         hipMemcpyDeviceToDevice, stream);

    const dim3 gridG(16, 4);
    for (int s = 0; s < S; ++s) {
        const int p = s & 1, q = p ^ 1;
        for (int l = 0; l < L; ++l) {
            const __bf16* inp = (l == 0) ? (xbf + (size_t)s * BH) : inp1;
            const float* hl  = hbuf + (size_t)p * L * BH + (size_t)l * BH;
            float*       hlo = hbuf + (size_t)q * L * BH + (size_t)l * BH;
            __bf16* nxt = (l == 0) ? inp1 : (topbf + (size_t)s * BH);
            gru_gates<<<gridG, 128, 0, stream>>>(
                inp, hl,
                Wrx + (size_t)l * HH, Wrh + (size_t)l * HH,
                Wzx + (size_t)l * HH, Wzh + (size_t)l * HH,
                Whx + (size_t)l * HH,
                br + l * H, bz + l * H, bh + l * H,
                rbuf, zbuf, xhbuf);
            gru_combine<<<gridG, 128, 0, stream>>>(
                hl, rbuf, zbuf, xhbuf, Whh + (size_t)l * HH, hlo, nxt);
        }
    }

    gru_logits<<<dim3(V / 64, (S * B) / 128), 128, 0, stream>>>(topbf, Wout, bout, out);

    (void)hipMemcpyAsync(out + (size_t)S * B * V, hbuf, L * BH * sizeof(float),
                         hipMemcpyDeviceToDevice, stream);
}
